// SelfAtt_70300024701784
// MI455X (gfx1250) — matrix operand order repaired and checked by reading, not compiled, not run on hardware
//
#include <hip/hip_runtime.h>

// ---------------------------------------------------------------------------
// Self-attention (N=4, C=128, HW=4096) for gfx1250 using v_wmma_f32_16x16x32_f16
//   Kernel 1: QKV = w @ x + b   (WMMA, f32->f16 convert on the fly)
//             outputs: Qt,Kt = [N][HW][C] f16 (transposed), V = [N][C][HW] f16
//   Kernel 2: flash-attention, one wave per 16-query tile.
//             S^T tiles  : D = A(K^T rows=keys) x B(Q cols=queries)
//             O'^T tiles : D = A(V rows=chan)   x B(P^T rows=keys)
// ---------------------------------------------------------------------------

typedef __attribute__((ext_vector_type(16))) _Float16 v16h;
typedef __attribute__((ext_vector_type(8)))  _Float16 v8h;
typedef __attribute__((ext_vector_type(2)))  _Float16 v2h;
typedef __attribute__((ext_vector_type(8)))  float    v8f;
typedef __attribute__((ext_vector_type(4)))  float    v4f;

union V16 {
    v16h v;
    v8h  h8[2];
    _Float16 e[16];
};

#define WMMA_F16(A, B, C) \
    __builtin_amdgcn_wmma_f32_16x16x32_f16(false, (A), false, (B), (short)0, (C), false, false)

static constexpr int Nb  = 4;
static constexpr int C   = 128;
static constexpr int HW  = 4096;

// A-fragment (16x32 f16) from a row of fp32 weights: elems 0..7 <- K=8h+0..7,
// elems 8..15 <- K=16+8h+0..7 (caller passes p already offset by 8*h floats).
__device__ inline V16 loadA_f32(const float* __restrict__ p) {
    V16 r;
    v4f a0 = *(const v4f*)(p);
    v4f a1 = *(const v4f*)(p + 4);
    v4f b0 = *(const v4f*)(p + 16);
    v4f b1 = *(const v4f*)(p + 20);
#pragma unroll
    for (int t = 0; t < 4; ++t) {
        r.e[t]      = (_Float16)a0[t];
        r.e[4 + t]  = (_Float16)a1[t];
        r.e[8 + t]  = (_Float16)b0[t];
        r.e[12 + t] = (_Float16)b1[t];
    }
    return r;
}

// ---------------------------------------------------------------------------
// Kernel 1: QKV projections. 1024 waves total; wave = (n, 16-pixel tile).
// ---------------------------------------------------------------------------
__global__ __launch_bounds__(256) void qkv_kernel(
    const float* __restrict__ x,
    const float* __restrict__ wq, const float* __restrict__ bq,
    const float* __restrict__ wk, const float* __restrict__ bk,
    const float* __restrict__ wv, const float* __restrict__ bv,
    _Float16* __restrict__ Qt, _Float16* __restrict__ Kt, _Float16* __restrict__ Vt)
{
    const int lane = threadIdx.x & 31;
    const int wave = threadIdx.x >> 5;
    const int tile = blockIdx.x * 8 + wave;      // 0..1023
    const int n    = tile >> 8;
    const int it   = tile & 255;
    const int i0   = it * 16;
    const int il   = lane & 15;
    const int i    = i0 + il;
    const int h    = lane >> 4;

    const float* xn = x + (size_t)n * C * HW;

    // B-fragments of x: lane col = pixel i, elem ev -> channel cc*32 + 8h + (ev&7) + 16*(ev>>3)
    V16 xb[4];
#pragma unroll
    for (int cc = 0; cc < 4; ++cc) {
#pragma unroll
        for (int ev = 0; ev < 16; ++ev) {
            const int c = cc * 32 + h * 8 + (ev & 7) + ((ev >> 3) << 4);
            xb[cc].e[ev] = (_Float16)xn[(size_t)c * HW + i];
        }
    }

    for (int dt = 0; dt < 8; ++dt) {
        v8f qacc = {}, kacc = {}, vacc = {};
        const int d = dt * 16 + il;              // A row (output channel) for this lane
#pragma unroll
        for (int cc = 0; cc < 4; ++cc) {
            const size_t wo = (size_t)d * C + cc * 32 + 8 * h;
            V16 aq = loadA_f32(wq + wo);
            V16 ak = loadA_f32(wk + wo);
            V16 av = loadA_f32(wv + wo);
            qacc = WMMA_F16(aq.v, xb[cc].v, qacc);
            kacc = WMMA_F16(ak.v, xb[cc].v, kacc);
            vacc = WMMA_F16(av.v, xb[cc].v, vacc);
        }
        // D layout: lane col = pixel i, VGPR j -> row d = dt*16 + j + 8h
        _Float16* qrow = Qt + ((size_t)n * HW + i) * C;
        _Float16* krow = Kt + ((size_t)n * HW + i) * C;
#pragma unroll
        for (int jp = 0; jp < 4; ++jp) {
            const int d0 = dt * 16 + 8 * h + 2 * jp;
            v2h qh, kh;
            qh[0] = (_Float16)(qacc[2 * jp]     + bq[d0]);
            qh[1] = (_Float16)(qacc[2 * jp + 1] + bq[d0 + 1]);
            kh[0] = (_Float16)(kacc[2 * jp]     + bk[d0]);
            kh[1] = (_Float16)(kacc[2 * jp + 1] + bk[d0 + 1]);
            *(v2h*)(qrow + d0) = qh;
            *(v2h*)(krow + d0) = kh;
        }
#pragma unroll
        for (int j = 0; j < 8; ++j) {
            const int dc = dt * 16 + j + 8 * h;
            Vt[((size_t)n * C + dc) * HW + i] = (_Float16)(vacc[j] + bv[dc]);
        }
    }
}

// ---------------------------------------------------------------------------
// Kernel 2: flash attention. 1024 waves; wave = (n, 16-query tile).
// ---------------------------------------------------------------------------
__global__ __launch_bounds__(256) void attn_kernel(
    const _Float16* __restrict__ Qt, const _Float16* __restrict__ Kt,
    const _Float16* __restrict__ Vt, const float* __restrict__ x,
    const float* __restrict__ gamma, float* __restrict__ out)
{
    const int lane = threadIdx.x & 31;
    const int wave = threadIdx.x >> 5;
    const int tile = blockIdx.x * 8 + wave;      // 0..1023
    const int n    = tile >> 8;
    const int it   = tile & 255;
    const int i0   = it * 16;
    const int il   = lane & 15;
    const int qi   = i0 + il;
    const int h    = lane >> 4;

    const _Float16* Qn = Qt + (size_t)n * HW * C;
    const _Float16* Kn = Kt + (size_t)n * HW * C;
    const _Float16* Vn = Vt + (size_t)n * C * HW;

    // Q B-fragments: lane col = query, 16 contiguous halfs starting at c = cc*32 + 16h
    v16h qb[4];
#pragma unroll
    for (int cc = 0; cc < 4; ++cc)
        qb[cc] = *(const v16h*)(Qn + (size_t)qi * C + cc * 32 + h * 16);

    v8f o[8];
#pragma unroll
    for (int ct = 0; ct < 8; ++ct) o[ct] = (v8f){};
    float m = -__builtin_inff();
    float l = 0.f;

    for (int jb = 0; jb < HW / 32; ++jb) {
        const int kbase = jb * 32;
        v8f sa = {}, sb = {};
        // S^T tiles: rows = keys (tile a: kbase+0..15, tile b: kbase+16..31)
#pragma unroll
        for (int cc = 0; cc < 4; ++cc) {
            const _Float16* ka = Kn + (size_t)(kbase + il) * C + cc * 32 + 16 * h;
            const _Float16* kb = ka + (size_t)16 * C;
            V16 fa, fb;
            fa.h8[0] = *(const v8h*)(ka);
            fa.h8[1] = *(const v8h*)(ka + 8);
            fb.h8[0] = *(const v8h*)(kb);
            fb.h8[1] = *(const v8h*)(kb + 8);
            sa = WMMA_F16(fa.v, qb[cc], sa);
            sb = WMMA_F16(fb.v, qb[cc], sb);
        }
        // --- online softmax (per lane: 16 of 32 keys local, partner has rest) ---
        float lm = fmaxf(sa[0], sb[0]);
#pragma unroll
        for (int j = 1; j < 8; ++j) lm = fmaxf(lm, fmaxf(sa[j], sb[j]));
        lm = fmaxf(lm, __shfl_xor(lm, 16, 32));
        const float mn    = fmaxf(m, lm);
        const float scale = __expf(m - mn);
        float pa[8], pb[8];
        float ls = 0.f;
#pragma unroll
        for (int j = 0; j < 8; ++j) {
            pa[j] = __expf(sa[j] - mn);
            pb[j] = __expf(sb[j] - mn);
            ls += pa[j] + pb[j];
        }
        ls += __shfl_xor(ls, 16, 32);
        l = l * scale + ls;
        m = mn;
#pragma unroll
        for (int ct = 0; ct < 8; ++ct) o[ct] *= scale;

        // --- P^T as B-fragment (rows = keys, cols = queries); one half-swap ---
        V16 p;
#pragma unroll
        for (int j = 0; j < 8; ++j) {
            const float lc = h ? pb[j] : pa[j];
            const float sw = h ? pa[j] : pb[j];
            const float xf = __shfl_xor(sw, 16, 32);
            p.e[j]     = (_Float16)(h ? xf : lc);   // elems 0..7  : keys 16h+0..7
            p.e[8 + j] = (_Float16)(h ? lc : xf);   // elems 8..15 : keys 16h+8..15
        }
        // --- O'^T += V_tile x P^T : A rows = channels ---
#pragma unroll
        for (int ct = 0; ct < 8; ++ct) {
            const _Float16* vp = Vn + (size_t)(ct * 16 + il) * HW + kbase + 16 * h;
            V16 fv;
            fv.h8[0] = *(const v8h*)(vp);
            fv.h8[1] = *(const v8h*)(vp + 8);
            o[ct] = WMMA_F16(fv.v, p.v, o[ct]);
        }
    }

    // Epilogue: out = gamma * O'/l + x.  D layout: lane col = query, row c = ct*16+j+8h
    const float g  = gamma[0];
    const float rl = g / l;
    const float* xn = x   + (size_t)n * C * HW;
    float*       on = out + (size_t)n * C * HW;
#pragma unroll
    for (int ct = 0; ct < 8; ++ct) {
#pragma unroll
        for (int j = 0; j < 8; ++j) {
            const int c = ct * 16 + j + 8 * h;
            const size_t idx = (size_t)c * HW + i0 + il;
            on[idx] = o[ct][j] * rl + xn[idx];
        }
    }
}

// ---------------------------------------------------------------------------
extern "C" void kernel_launch(void* const* d_in, const int* in_sizes, int n_in,
                              void* d_out, int out_size, void* d_ws, size_t ws_size,
                              hipStream_t stream)
{
    const float* x     = (const float*)d_in[0];
    const float* wq    = (const float*)d_in[1];
    const float* bq    = (const float*)d_in[2];
    const float* wk    = (const float*)d_in[3];
    const float* bk    = (const float*)d_in[4];
    const float* wv    = (const float*)d_in[5];
    const float* bv    = (const float*)d_in[6];
    const float* gamma = (const float*)d_in[7];
    float* out = (float*)d_out;

    _Float16* Qt = (_Float16*)d_ws;                     // 4 MB
    _Float16* Kt = Qt + (size_t)Nb * HW * C;            // 4 MB
    _Float16* Vt = Kt + (size_t)Nb * HW * C;            // 4 MB

    // 1024 waves each: 128 blocks x 256 threads (8 waves/block)
    qkv_kernel<<<128, 256, 0, stream>>>(x, wq, bq, wk, bk, wv, bv, Qt, Kt, Vt);
    attn_kernel<<<128, 256, 0, stream>>>(Qt, Kt, Vt, x, gamma, out);
}
